// NanoGPT_41497974014416
// MI455X (gfx1250) — matrix operand order repaired and checked by reading, not compiled, not run on hardware
//
#include <hip/hip_runtime.h>
#include <hip/hip_bf16.h>

// ---------- CDNA5 WMMA types ----------
typedef __bf16          v16bf  __attribute__((ext_vector_type(16)));
typedef float           v8f    __attribute__((ext_vector_type(8)));
typedef unsigned short  u16x8  __attribute__((ext_vector_type(8)));
typedef unsigned int    v4u    __attribute__((ext_vector_type(4)));
typedef int             v4i    __attribute__((ext_vector_type(4)));
typedef int             v8i    __attribute__((ext_vector_type(8)));

union Frag {
    v16bf bf;
    u16x8 us[2];
};

#if __has_builtin(__builtin_amdgcn_tensor_load_to_lds)
#define HAVE_TDM 1
#define ASYNC_IF "0x4"          // in-flight next-stage async ops (A tile only)
#else
#define HAVE_TDM 0
#define ASYNC_IF "0x6"          // A (4) + B (2) per-lane async
#endif

__device__ __forceinline__ unsigned short f2bf(float f) {
    unsigned int u = __float_as_uint(f);
    u += 0x7FFFu + ((u >> 16) & 1u);      // round-to-nearest-even
    return (unsigned short)(u >> 16);
}
__device__ __forceinline__ unsigned int pack2(float lo, float hi) {
    return (unsigned int)f2bf(lo) | ((unsigned int)f2bf(hi) << 16);
}
// low 32 bits of a generic pointer to __shared__ = wave-relative LDS byte address
__device__ __forceinline__ unsigned lds_addr(const void* p) {
    return (unsigned)(unsigned long long)p;
}

// ---- TDM: one descriptor moves a 64x64 bf16 tile (row stride K) into LDS
//      with 4-DWORD padding per 32-DWORD row -> 72-halfword LDS stride.
__device__ __forceinline__ void tdm_load_tile_b(
    const unsigned short* gsrc, unsigned ldsb, int N, int K)
{
#if HAVE_TDM
    unsigned long long ga = (unsigned long long)gsrc;
    v4u g0;
    g0[0] = 1u;                                            // count=1, user mode
    g0[1] = ldsb;                                          // lds_addr (bytes)
    g0[2] = (unsigned)(ga & 0xFFFFFFFFull);                // global_addr[31:0]
    g0[3] = (unsigned)((ga >> 32) & 0x01FFFFFFull)         // global_addr[56:32]
          | (2u << 30);                                    // type=2 (image)
    v8i g1;
    g1[0] = (int)((1u << 16)      // data_size = 2 bytes
                | (1u << 20)      // pad_enable
                | (4u << 22)      // pad_interval: 32 DWORDs
                | (3u << 25));    // pad_amount: 4 DWORDs
    g1[1] = (int)(((unsigned)K & 0xFFFFu) << 16);                          // tensor_dim0 lo
    g1[2] = (int)((((unsigned)K >> 16) & 0xFFFFu)
                | (((unsigned)N & 0xFFFFu) << 16));                        // dim0 hi | dim1 lo
    g1[3] = (int)((((unsigned)N >> 16) & 0xFFFFu) | (64u << 16));          // dim1 hi | tile_dim0
    g1[4] = 64;                                                            // tile_dim1 (tile_dim2=0)
    g1[5] = (int)(unsigned)K;                                              // tensor_dim0_stride lo
    g1[6] = 0;                                                             // stride hi | dim1_stride lo
    g1[7] = 0;
    const v4i z4 = {0, 0, 0, 0};
#if __clang_major__ >= 23
    const v8i z8 = {0, 0, 0, 0, 0, 0, 0, 0};
    __builtin_amdgcn_tensor_load_to_lds(g0, g1, z4, z4, z8, 0);
#else
    __builtin_amdgcn_tensor_load_to_lds(g0, g1, z4, z4, 0);
#endif
#else
    (void)gsrc; (void)ldsb; (void)N; (void)K;
#endif
}

#if HAVE_TDM
#define WAIT_TENSOR(n) __builtin_amdgcn_s_wait_tensorcnt(n)
#else
#define WAIT_TENSOR(n)
#endif

// ==========================================================================
// Tiled bf16 WMMA GEMM:  C = A(MxK) * B^T  with B stored N x K row-major.
// A tile: per-lane GLOBAL_LOAD_ASYNC_TO_LDS_B128 (ASYNCcnt).
// B tile: TDM TENSOR_LOAD_TO_LDS, one descriptor per stage (TENSORcnt).
// Double-buffered LDS, software-pipelined, last iteration peeled.
// ==========================================================================
static constexpr int BM = 128, BN = 64, BK = 64;
static constexpr int LDT = 72;           // halfword row stride (144B, 16B-aligned chunks)

template<bool BIAS, bool RELU, bool RESID, bool OUTBF16>
__global__ __launch_bounds__(256) void gemm_bf16_wmma(
    const unsigned short* __restrict__ Ag,   // M x K row-major bf16
    const unsigned short* __restrict__ Bg,   // N x K row-major bf16
    const float* __restrict__ bias,
    const float* __restrict__ Rg,            // residual, f32, MxN
    void* __restrict__ Cout,                 // f32 or bf16 per OUTBF16
    int M, int N, int K,
    long long sA, long long sB, long long sC)
{
    __shared__ __align__(16) unsigned short As[2][BM * LDT];
    __shared__ __align__(16) unsigned short Bs[2][BN * LDT];

    const long long bz = blockIdx.z;
    Ag += bz * sA;
    Bg += bz * sB;
    const long long coff = bz * sC;

    const int m0   = blockIdx.y * BM;
    const int n0   = blockIdx.x * BN;
    const int tid  = threadIdx.x;
    const int lane = tid & 31;
    const int wave = tid >> 5;
    const int wr   = wave & 3;    // 4 waves along M (4*32 = 128)
    const int wc   = wave >> 2;   // 2 waves along N (2*32 = 64)
    const int h    = lane >> 4;
    const int ml   = lane & 15;

    // per-thread staging coordinates (8-halfword = 16B chunks)
    const int arow = tid >> 3, achk = tid & 7;    // A: 128 rows x 8 chunks, 4/thread

    auto issue_stage = [&](int buf, int k0) {
        if (wave == 0)
            tdm_load_tile_b(Bg + (long long)n0 * K + k0, lds_addr(&Bs[buf][0]), N, K);
#if !HAVE_TDM
        #pragma unroll
        for (int i = 0; i < 2; ++i) {
            int row = arow + i * 32;
            unsigned long long g = (unsigned long long)(Bg + (long long)(n0 + row) * K + k0 + achk * 8);
            unsigned l = lds_addr(&Bs[buf][row * LDT + achk * 8]);
            asm volatile("global_load_async_to_lds_b128 %0, %1, off"
                         :: "v"(l), "v"(g) : "memory");
        }
#endif
        #pragma unroll
        for (int i = 0; i < 4; ++i) {
            int row = arow + i * 32;
            unsigned long long g = (unsigned long long)(Ag + (long long)(m0 + row) * K + k0 + achk * 8);
            unsigned l = lds_addr(&As[buf][row * LDT + achk * 8]);
            asm volatile("global_load_async_to_lds_b128 %0, %1, off"
                         :: "v"(l), "v"(g) : "memory");
        }
    };

    const v8f vzero = {0.f, 0.f, 0.f, 0.f, 0.f, 0.f, 0.f, 0.f};
    v8f acc[2][2];
    #pragma unroll
    for (int i = 0; i < 2; ++i)
        #pragma unroll
        for (int j = 0; j < 2; ++j) acc[i][j] = vzero;

    auto compute_stage = [&](int cur) {
        #pragma unroll
        for (int ks = 0; ks < 2; ++ks) {        // two 16x16x32 K-steps per stage
            const int kk = ks * 32;
            Frag a[2], b[2];
            #pragma unroll
            for (int i = 0; i < 2; ++i) {
                const unsigned short* p = &As[cur][(wr * 32 + i * 16 + ml) * LDT + kk];
                a[i].us[0] = *reinterpret_cast<const u16x8*>(p + 8 * h);       // K = 8h..
                a[i].us[1] = *reinterpret_cast<const u16x8*>(p + 16 + 8 * h);  // K = 16+8h..
            }
            #pragma unroll
            for (int j = 0; j < 2; ++j) {
                const unsigned short* p = &Bs[cur][(wc * 32 + j * 16 + ml) * LDT + kk];
                b[j].us[0] = *reinterpret_cast<const u16x8*>(p + 8 * h);       // K = 8h..
                b[j].us[1] = *reinterpret_cast<const u16x8*>(p + 16 + 8 * h);
            }
            #pragma unroll
            for (int i = 0; i < 2; ++i)
                #pragma unroll
                for (int j = 0; j < 2; ++j)
                    acc[i][j] = __builtin_amdgcn_wmma_f32_16x16x32_bf16(
                        false, a[i].bf, false, b[j].bf, (short)0, acc[i][j], false, false);
        }
    };

    const int nk = K >> 6;                 // K / 64 (>= 2 for all call sites)
    issue_stage(0, 0);

    // steady state: issue stage kt+1, wait stage kt resident, compute stage kt
    for (int kt = 0; kt < nk - 1; ++kt) {
        const int cur = kt & 1;
        issue_stage(cur ^ 1, (kt + 1) << 6);
        // async completes in order: <= in-flight(next) means current stage resident
        asm volatile("s_wait_asynccnt " ASYNC_IF ::: "memory");
        WAIT_TENSOR(1);
        __syncthreads();
        compute_stage(cur);
        __syncthreads();
    }
    // peeled final stage
    asm volatile("s_wait_asynccnt 0x0" ::: "memory");
    WAIT_TENSOR(0);
    __syncthreads();
    compute_stage((nk - 1) & 1);

    // ---- epilogue: C layout lane%16 = N, vgpr r + 8*(lane/16) = M
    #pragma unroll
    for (int i = 0; i < 2; ++i) {
        const int rowbase = m0 + wr * 32 + i * 16 + h * 8;
        #pragma unroll
        for (int j = 0; j < 2; ++j) {
            const int col = n0 + wc * 32 + j * 16 + ml;
            float bv = 0.0f;
            if (BIAS) bv = bias[col];
            const long long off0 = coff + (long long)rowbase * N + col;
            #pragma unroll
            for (int r = 0; r < 8; ++r) {
                const long long off = off0 + (long long)r * N;
                float v = acc[i][j][r];
                if (BIAS)  v += bv;
                if (RELU)  v = fmaxf(v, 0.0f);
                if (RESID) v += Rg[off];
                if (OUTBF16) ((unsigned short*)Cout)[off] = f2bf(v);
                else         ((float*)Cout)[off] = v;
            }
        }
    }
}

// ---------- flat f32 -> bf16 convert (4 elems/thread, float4 in, uint2 out) ----------
__global__ __launch_bounds__(256) void cvt_bf16_kernel(
    const float* __restrict__ src, unsigned short* __restrict__ dst)
{
    long long i = ((long long)blockIdx.x * 256 + threadIdx.x) * 4;
    const float4 f = *reinterpret_cast<const float4*>(src + i);
    uint2 o;
    o.x = pack2(f.x, f.y);
    o.y = pack2(f.z, f.w);
    *reinterpret_cast<uint2*>(dst + i) = o;
}

// ---------- transpose + convert: dst[c*R + r] = bf16(src[r*C + c]) ----------
__global__ __launch_bounds__(256) void transpose_cvt_kernel(
    const float* __restrict__ src, unsigned short* __restrict__ dst,
    int R, int C, long long sSrc, long long sDst)
{
    __shared__ float t[32][33];
    src += (long long)blockIdx.z * sSrc;
    dst += (long long)blockIdx.z * sDst;
    const int r0 = blockIdx.y * 32, c0 = blockIdx.x * 32;
    const int tx = threadIdx.x & 31, ty = threadIdx.x >> 5;
    #pragma unroll
    for (int i = 0; i < 32; i += 8)
        t[ty + i][tx] = src[(long long)(r0 + ty + i) * C + c0 + tx];
    __syncthreads();
    #pragma unroll
    for (int i = 0; i < 32; i += 8)
        dst[(long long)(c0 + ty + i) * R + r0 + tx] = f2bf(t[tx][ty + i]);
}

// ---------- embedding gather ----------
__global__ __launch_bounds__(256) void embed_kernel(
    const int* __restrict__ idx, const float* __restrict__ enc,
    float* __restrict__ z, int D)
{
    long long row = blockIdx.x;
    long long src = (long long)idx[row] * D;
    for (int i = threadIdx.x; i < D; i += 256)
        z[row * D + i] = enc[src + i];
}

// ---------- LayerNorm (faithful buggy form: (x - mu/sqrt(var))*g + b, ddof=1), bf16 out ----------
__global__ __launch_bounds__(256) void ln_bf16_kernel(
    const float* __restrict__ x, const float* __restrict__ g,
    const float* __restrict__ b, unsigned short* __restrict__ y, int D)
{
    __shared__ float sh[256], sh2[256];
    long long base = (long long)blockIdx.x * D;
    float s = 0.f, s2 = 0.f;
    for (int i = threadIdx.x; i < D; i += 256) {
        float v = x[base + i];
        s += v; s2 += v * v;
    }
    sh[threadIdx.x] = s; sh2[threadIdx.x] = s2;
    __syncthreads();
    for (int st = 128; st > 0; st >>= 1) {
        if (threadIdx.x < st) {
            sh[threadIdx.x]  += sh[threadIdx.x + st];
            sh2[threadIdx.x] += sh2[threadIdx.x + st];
        }
        __syncthreads();
    }
    float mu    = sh[0] / (float)D;
    float var   = (sh2[0] - (float)D * mu * mu) / (float)(D - 1);
    float shift = mu * rsqrtf(var);
    for (int i = threadIdx.x; i < D; i += 256)
        y[base + i] = f2bf((x[base + i] - shift) * g[i] + b[i]);
}

// ---------- scaled row softmax, f32 in -> bf16 out ----------
__global__ __launch_bounds__(256) void softmax_bf16_kernel(
    const float* __restrict__ s, unsigned short* __restrict__ out,
    int n, float scale)
{
    __shared__ float sh[256];
    long long base = (long long)blockIdx.x * n;
    float m = -1e30f;
    for (int i = threadIdx.x; i < n; i += 256)
        m = fmaxf(m, s[base + i] * scale);
    sh[threadIdx.x] = m; __syncthreads();
    for (int st = 128; st > 0; st >>= 1) {
        if (threadIdx.x < st) sh[threadIdx.x] = fmaxf(sh[threadIdx.x], sh[threadIdx.x + st]);
        __syncthreads();
    }
    m = sh[0]; __syncthreads();
    float sum = 0.f;
    for (int i = threadIdx.x; i < n; i += 256)
        sum += expf(s[base + i] * scale - m);
    sh[threadIdx.x] = sum; __syncthreads();
    for (int st = 128; st > 0; st >>= 1) {
        if (threadIdx.x < st) sh[threadIdx.x] += sh[threadIdx.x + st];
        __syncthreads();
    }
    float inv = 1.0f / sh[0];
    for (int i = threadIdx.x; i < n; i += 256)
        out[base + i] = f2bf(expf(s[base + i] * scale - m) * inv);
}

// ---------- loss ----------
__global__ void loss_zero(float* p) {
    if (threadIdx.x == 0 && blockIdx.x == 0) *p = 0.0f;
}

__global__ __launch_bounds__(256) void nll_kernel(
    const float* __restrict__ logits, const int* __restrict__ tgt,
    float* __restrict__ loss, int V, float invn)
{
    __shared__ float sh[256];
    long long base = (long long)blockIdx.x * V;
    float m = -1e30f;
    for (int i = threadIdx.x; i < V; i += 256)
        m = fmaxf(m, logits[base + i]);
    sh[threadIdx.x] = m; __syncthreads();
    for (int st = 128; st > 0; st >>= 1) {
        if (threadIdx.x < st) sh[threadIdx.x] = fmaxf(sh[threadIdx.x], sh[threadIdx.x + st]);
        __syncthreads();
    }
    m = sh[0]; __syncthreads();
    float sum = 0.f;
    for (int i = threadIdx.x; i < V; i += 256)
        sum += expf(logits[base + i] - m);
    sh[threadIdx.x] = sum; __syncthreads();
    for (int st = 128; st > 0; st >>= 1) {
        if (threadIdx.x < st) sh[threadIdx.x] += sh[threadIdx.x + st];
        __syncthreads();
    }
    if (threadIdx.x == 0) {
        float lse = m + logf(sh[0]);
        float nll = lse - logits[base + tgt[blockIdx.x]];
        atomicAdd(loss, nll * invn);
    }
}

// ==========================================================================
// driver
// ==========================================================================
extern "C" void kernel_launch(void* const* d_in, const int* in_sizes, int n_in,
                              void* d_out, int out_size, void* d_ws, size_t ws_size,
                              hipStream_t stream) {
    (void)in_sizes; (void)n_in; (void)out_size; (void)ws_size;

    const int*   xi  = (const int*)  d_in[0];
    const int*   yi  = (const int*)  d_in[1];
    const float* enc = (const float*)d_in[2];
    const float* g1  = (const float*)d_in[3];
    const float* b1  = (const float*)d_in[4];
    const float* wq  = (const float*)d_in[5];
    const float* wk  = (const float*)d_in[6];
    const float* wv  = (const float*)d_in[7];
    const float* g2  = (const float*)d_in[8];
    const float* b2  = (const float*)d_in[9];
    const float* w1  = (const float*)d_in[10];
    const float* bb1 = (const float*)d_in[11];
    const float* w2  = (const float*)d_in[12];
    const float* bb2 = (const float*)d_in[13];

    constexpr int Bb = 4, T = 1024, D = 1024, V = 32000, L = 6;
    constexpr int BT = Bb * T;
    const long long TD = (long long)T * D;
    const long long TT = (long long)T * T;
    const long long DD = (long long)D * D;

    // ---- workspace carve-up
    char* wp = (char*)d_ws;
    auto allocF = [&](long long n) { float* p = (float*)wp; wp += n * sizeof(float); return p; };
    auto allocH = [&](long long n) { unsigned short* p = (unsigned short*)wp; wp += n * sizeof(unsigned short); return p; };

    float* z  = allocF((long long)BT * D);   // residual stream (f32)
    float* s  = allocF(TT * Bb);             // attention scores (f32)
    float* o  = allocF((long long)BT * D);   // v (f32) then attn-out (f32)
    unsigned short* yb  = allocH((long long)BT * D);  // ln out (bf16)
    unsigned short* qb  = allocH((long long)BT * D);  // q bf16, later ffn hidden
    unsigned short* kb  = allocH((long long)BT * D);
    unsigned short* vtb = allocH((long long)BT * D);  // v transposed per batch (D x T)
    unsigned short* ab  = allocH(TT * Bb);            // softmax probs bf16
    unsigned short* zb  = allocH((long long)BT * D);  // z bf16 for decoder
    unsigned short* wqb = allocH(DD);
    unsigned short* wkb = allocH(DD);
    unsigned short* wvb = allocH(DD);
    unsigned short* w1b = allocH(DD);
    unsigned short* w2b = allocH(DD);
    unsigned short* encb = allocH((long long)V * D);

    float* logits = (float*)d_out;                 // (BT, V)
    float* loss   = logits + (long long)BT * V;    // scalar tail

    dim3 blk(256);

    // ---- one-time (per call) weight transpose-convert: (K,N) f32 -> (N,K) bf16
    const dim3 gW(D / 32, D / 32, 1);
    transpose_cvt_kernel<<<gW, blk, 0, stream>>>(wq, wqb, D, D, 0, 0);
    transpose_cvt_kernel<<<gW, blk, 0, stream>>>(wk, wkb, D, D, 0, 0);
    transpose_cvt_kernel<<<gW, blk, 0, stream>>>(wv, wvb, D, D, 0, 0);
    transpose_cvt_kernel<<<gW, blk, 0, stream>>>(w1, w1b, D, D, 0, 0);
    transpose_cvt_kernel<<<gW, blk, 0, stream>>>(w2, w2b, D, D, 0, 0);
    // encoder is already (V, D) = (N, K): flat convert
    cvt_bf16_kernel<<<(int)(((long long)V * D) / 1024), blk, 0, stream>>>(enc, encb);

    embed_kernel<<<BT, blk, 0, stream>>>(xi, enc, z, D);

    const dim3 gDD(D / BN, BT / BM, 1);
    const dim3 gQK(T / BN, T / BM, Bb);
    const dim3 gAV(D / BN, T / BM, Bb);
    const dim3 gVT(D / 32, T / 32, Bb);

    for (int l = 0; l < L; ++l) {
        ln_bf16_kernel<<<BT, blk, 0, stream>>>(z, g1, b1, yb, D);

        // q, k in bf16 directly (only consumed by WMMA)
        gemm_bf16_wmma<false, false, false, true><<<gDD, blk, 0, stream>>>(
            yb, wqb, nullptr, nullptr, qb, BT, D, D, 0, 0, 0);
        gemm_bf16_wmma<false, false, false, true><<<gDD, blk, 0, stream>>>(
            yb, wkb, nullptr, nullptr, kb, BT, D, D, 0, 0, 0);
        // v in f32, then per-batch transpose to (D, T) bf16
        gemm_bf16_wmma<false, false, false, false><<<gDD, blk, 0, stream>>>(
            yb, wvb, nullptr, nullptr, o, BT, D, D, 0, 0, 0);
        transpose_cvt_kernel<<<gVT, blk, 0, stream>>>(o, vtb, T, D, TD, TD);

        // S = Q * K^T (per batch): A = q (TxD), B = k (TxD, N x K layout)
        gemm_bf16_wmma<false, false, false, false><<<gQK, blk, 0, stream>>>(
            qb, kb, nullptr, nullptr, s, T, T, D, TD, TD, TT);
        softmax_bf16_kernel<<<BT, blk, 0, stream>>>(s, ab, T, 1.0f / 32.0f);
        // O = A * V: B = v^T (DxT, N x K layout)
        gemm_bf16_wmma<false, false, false, false><<<gAV, blk, 0, stream>>>(
            ab, vtb, nullptr, nullptr, o, T, D, T, TT, TD, TD);

        ln_bf16_kernel<<<BT, blk, 0, stream>>>(o, g2, b2, yb, D);

        // h = relu(y*w1 + bb1), bf16 (into qb, no longer needed)
        gemm_bf16_wmma<true, true, false, true><<<gDD, blk, 0, stream>>>(
            yb, w1b, bb1, nullptr, qb, BT, D, D, 0, 0, 0);
        // z = z + (h*w2 + bb2), f32
        gemm_bf16_wmma<true, false, true, false><<<gDD, blk, 0, stream>>>(
            qb, w2b, bb2, z, z, BT, D, D, 0, 0, 0);
    }

    // decoder: logits = z * enc^T
    cvt_bf16_kernel<<<(int)(((long long)BT * D) / 1024), blk, 0, stream>>>(z, zb);
    const dim3 gLG(V / BN, BT / BM, 1);
    gemm_bf16_wmma<false, false, false, false><<<gLG, blk, 0, stream>>>(
        zb, encb, nullptr, nullptr, logits, BT, V, D, 0, 0, 0);

    loss_zero<<<1, 32, 0, stream>>>(loss);
    nll_kernel<<<BT, blk, 0, stream>>>(logits, yi, loss, V, 1.0f / (float)BT);
}
